// GNNpool_20452634263687
// MI455X (gfx1250) — matrix operand order repaired and checked by reading, not compiled, not run on hardware
//
#include <hip/hip_runtime.h>
#include <hip/hip_bf16.h>

// ---------------- problem constants ----------------
#define NN      10000
#define EE      320000
#define ETOT    (EE + NN)     // with self loops
#define DIN     768
#define CH      512
#define MLPH    256
#define NEGS    0.2f

typedef __attribute__((ext_vector_type(16))) __bf16 v16bf;
typedef __attribute__((ext_vector_type(8)))  __bf16 bfx8;
typedef __attribute__((ext_vector_type(8)))  float  v8f;
typedef __attribute__((ext_vector_type(4)))  float  f4;

__device__ __forceinline__ float silu_f(float x) { return x / (1.f + __expf(-x)); }

// ordered-uint mapping so unsigned atomicMax implements float max (handles negatives)
__device__ __forceinline__ unsigned f2ord(float f) {
    unsigned u = __float_as_uint(f);
    return (u >> 31) ? ~u : (u | 0x80000000u);
}
__device__ __forceinline__ float ord2f(unsigned u) {
    return (u >> 31) ? __uint_as_float(u & 0x7fffffffu) : __uint_as_float(~u);
}

// ---------------- utility kernels ----------------
__global__ void copy_nt_f4(const f4* __restrict__ s, f4* __restrict__ d, int n4) {
    int i = blockIdx.x * blockDim.x + threadIdx.x;
    if (i < n4) {
        f4 v = __builtin_nontemporal_load(s + i);
        __builtin_nontemporal_store(v, d + i);
    }
}

__global__ void cvt_f32_bf16(const float* __restrict__ s, __bf16* __restrict__ d, int n) {
    int i = blockIdx.x * blockDim.x + threadIdx.x;
    if (i < n) d[i] = (__bf16)s[i];
}

__global__ void fill_f32(float* p, float v, int n) {
    int i = blockIdx.x * blockDim.x + threadIdx.x;
    if (i < n) p[i] = v;
}
__global__ void fill_u32(unsigned* p, unsigned v, int n) {
    int i = blockIdx.x * blockDim.x + threadIdx.x;
    if (i < n) p[i] = v;
}

// self-loop attr: mean of incoming edge attrs per destination
__global__ void selfloop_stats(const int* __restrict__ dst, const float* __restrict__ ea,
                               float* cnt, float* sum, int ne) {
    int e = blockIdx.x * blockDim.x + threadIdx.x;
    if (e < ne) {
        int d = dst[e];
        atomicAdd(&cnt[d], 1.f);
        atomicAdd(&sum[d], ea[e]);
    }
}

// dot of two 512-vectors -> scalar (edge attention scalar)
__global__ void dot512(const float* __restrict__ a, const float* __restrict__ b, float* out) {
    __shared__ float red[256];
    int t = threadIdx.x;
    red[t] = a[t] * b[t] + a[t + 256] * b[t + 256];
    __syncthreads();
    for (int s = 128; s; s >>= 1) {
        if (t < s) red[t] += red[t + s];
        __syncthreads();
    }
    if (t == 0) *out = red[0];
}

// per-node alpha_src/alpha_dst: wave per node, lane-strided, shfl reduce (wave32)
__global__ void node_dots(const float* __restrict__ h, const float* __restrict__ aw,
                          const float* __restrict__ dw, float* __restrict__ ao,
                          float* __restrict__ dozo, int n, int c) {
    int wid  = (blockIdx.x * blockDim.x + threadIdx.x) >> 5;
    int lane = threadIdx.x & 31;
    if (wid >= n) return;
    const float* row = h + (size_t)wid * c;
    float s1 = 0.f, s2 = 0.f;
    for (int k = lane; k < c; k += 32) {
        float v = row[k];
        s1 += v * aw[k];
        s2 += v * dw[k];
    }
    for (int o = 16; o; o >>= 1) {
        s1 += __shfl_down(s1, o);
        s2 += __shfl_down(s2, o);
    }
    if (lane == 0) { ao[wid] = s1; dozo[wid] = s2; }
}

// alpha = leaky_relu(as[src]+ad[dst]+ea*esc); segment-max via ordered atomicMax
__global__ void alpha_max(const int* __restrict__ src, const int* __restrict__ dst,
                          const float* __restrict__ ea, const float* __restrict__ lsum,
                          const float* __restrict__ lcnt, const float* __restrict__ ao,
                          const float* __restrict__ dozo, const float* __restrict__ esc,
                          float* __restrict__ alpha, unsigned* __restrict__ amax) {
    int e = blockIdx.x * blockDim.x + threadIdx.x;
    if (e >= ETOT) return;
    int s, d; float a;
    if (e < EE) { s = src[e]; d = dst[e]; a = ea[e]; }
    else        { s = d = e - EE; a = lsum[s] / fmaxf(lcnt[s], 1.f); }
    float x = ao[s] + dozo[d] + a * (*esc);
    x = (x > 0.f) ? x : NEGS * x;
    alpha[e] = x;
    atomicMax(&amax[d], f2ord(x));
}

__global__ void ex_denom(const int* __restrict__ dst, float* __restrict__ alpha,
                         const unsigned* __restrict__ amax, float* __restrict__ denom) {
    int e = blockIdx.x * blockDim.x + threadIdx.x;
    if (e >= ETOT) return;
    int d = (e < EE) ? dst[e] : e - EE;
    float ex = __expf(alpha[e] - ord2f(amax[d]));
    alpha[e] = ex;
    atomicAdd(&denom[d], ex);
}

__global__ void finalize_w(const int* __restrict__ dst, float* __restrict__ alpha,
                           const float* __restrict__ denom) {
    int e = blockIdx.x * blockDim.x + threadIdx.x;
    if (e >= ETOT) return;
    int d = (e < EE) ? dst[e] : e - EE;
    alpha[e] = alpha[e] / (denom[d] + 1e-16f);
}

// agg[dst] += w_e * h[src]; one block per edge, 256 threads x float2
__global__ void __launch_bounds__(256)
scatter_agg(const int* __restrict__ src, const int* __restrict__ dst,
            const float* __restrict__ w, const float* __restrict__ h,
            float* __restrict__ agg) {
    int e = blockIdx.x;
    int s, d;
    if (e < EE) { s = src[e]; d = dst[e]; }
    else        { s = d = e - EE; }
    float we = w[e];
    const float2* hs = (const float2*)(h + (size_t)s * CH);
    float* ag = agg + (size_t)d * CH;
    int t = threadIdx.x;             // CH/2 == 256
    float2 v = hs[t];
    atomicAdd(&ag[2 * t],     we * v.x);
    atomicAdd(&ag[2 * t + 1], we * v.y);
}

// out_bf16 = bf16(silu(agg + bias)); c must be power of two
__global__ void bias_silu_bf16(const float* __restrict__ agg, const float* __restrict__ bias,
                               __bf16* __restrict__ outb, int total, int cmask) {
    int i = blockIdx.x * blockDim.x + threadIdx.x;
    if (i >= total) return;
    float v = agg[i] + bias[i & cmask];
    outb[i] = (__bf16)silu_f(v);
}

// ---------------- WMMA bf16 GEMM: C[M,Nout] = X[M,K] * W[Nout,K]^T ----------------
// block tile 128x128, 8 waves -> each wave a 32x64 (2x4) quad of 16x16x32 WMMA tiles.
// A tile is DMA'd into LDS with GLOBAL_LOAD_ASYNC_TO_LDS_B128 (ASYNCcnt path);
// B tile is transposed through registers into LDS ([K][N], N contiguous).
__global__ void __launch_bounds__(256)
gemm_bf16_wmma(const __bf16* __restrict__ Xb, const __bf16* __restrict__ Wb,
               float* __restrict__ outF, __bf16* __restrict__ outB,
               const float* __restrict__ bias, int M, int Kd, int Nout, int act) {
    __shared__ __bf16 ldsA[128][40];     // 128x32 tile, rows padded to 40 halves
    __shared__ __bf16 ldsBT[32][140];    // 32x128 tile transposed, rows padded to 140

    const int tid  = threadIdx.x;
    const int lane = tid & 31;
    const int wid  = tid >> 5;
    const int wm   = wid & 3;            // 4 waves along M (x32 rows)
    const int wn   = wid >> 2;           // 2 waves along N (x64 cols)
    const int m0   = blockIdx.x * 128;
    const int n0   = blockIdx.y * 128;

    v8f acc[2][4] = {};

    // A-tile async-copy assignment: thread t -> row t/2, 32B chunk (t&1)
    const int arow = tid >> 1;           // 0..127
    const int acol = (tid & 1) * 16;     // 0 or 16
    const int rr   = (m0 + arow < M) ? (m0 + arow) : (M - 1);   // clamp; junk rows discarded
    const unsigned ldsA_dst = (unsigned)(uintptr_t)&ldsA[arow][acol];

    // B-tile assignment: thread t -> W row t/2, 16 cols starting at (t&1)*16
    const int bn = tid >> 1;             // 0..127 (Nout multiple of 128 -> no guard)
    const int bk = (tid & 1) * 16;

    for (int kt = 0; kt < Kd; kt += 32) {
        // ---- B tile: global -> regs (transpose happens on the LDS store) ----
        const __bf16* wg = Wb + (size_t)(n0 + bn) * Kd + kt + bk;
        bfx8 b0 = *(const bfx8*)wg;
        bfx8 b1 = *(const bfx8*)(wg + 8);
        if (kt + 32 < Kd) __builtin_prefetch(wg + 32, 0, 0);   // global_prefetch_b8

        __syncthreads();   // previous iteration finished reading LDS

        // ---- A tile: async DMA straight into LDS (no VGPR round trip) ----
        {
            unsigned long long ga =
                (unsigned long long)(uintptr_t)(Xb + (size_t)rr * Kd + kt + acol);
            asm volatile("global_load_async_to_lds_b128 %0, %1, off"
                         :: "v"(ldsA_dst), "v"(ga) : "memory");
            asm volatile("global_load_async_to_lds_b128 %0, %1, off offset:16"
                         :: "v"(ldsA_dst), "v"(ga) : "memory");
        }

        // ---- B tile: transposed scalar stores into LDS ----
        #pragma unroll
        for (int i = 0; i < 8; i++) ldsBT[bk + i][bn]     = b0[i];
        #pragma unroll
        for (int i = 0; i < 8; i++) ldsBT[bk + 8 + i][bn] = b1[i];

        asm volatile("s_wait_asynccnt 0x0" ::: "memory");      // our DMA landed
        __syncthreads();                                       // all waves' writes visible

        // ---- fragments + 8 WMMA per wave ----
        const int fr = lane & 15;
        const int kb = (lane >> 4) * 8;
        #pragma unroll
        for (int mi = 0; mi < 2; mi++) {
            v16bf afr;
            const __bf16* ar = &ldsA[wm * 32 + mi * 16 + fr][0];
            #pragma unroll
            for (int j = 0; j < 8; j++) { afr[j] = ar[kb + j]; afr[8 + j] = ar[16 + kb + j]; }
            #pragma unroll
            for (int ni = 0; ni < 4; ni++) {
                v16bf bfr;
                const __bf16* br = &ldsBT[0][wn * 64 + ni * 16 + fr];
                #pragma unroll
                for (int j = 0; j < 8; j++) { bfr[j] = br[(kb + j) * 140]; bfr[8 + j] = br[(16 + kb + j) * 140]; }
                acc[mi][ni] = __builtin_amdgcn_wmma_f32_16x16x32_bf16(
                    false, afr, false, bfr, (short)0, acc[mi][ni], false, false);
            }
        }
    }

    // ---- epilogue: C layout -> lane<16: M=v, N=lane ; lane>=16: M=8+v, N=lane-16 ----
    const int cb = n0 + wn * 64 + (lane & 15);
    const int rb = m0 + wm * 32 + ((lane >> 4) << 3);
    #pragma unroll
    for (int mi = 0; mi < 2; mi++) {
        #pragma unroll
        for (int ni = 0; ni < 4; ni++) {
            int cc = cb + ni * 16;
            #pragma unroll
            for (int v = 0; v < 8; v++) {
                int r = rb + mi * 16 + v;
                if (r < M) {
                    float val = acc[mi][ni][v];
                    if (bias) val += bias[cc];
                    if (act)  val = silu_f(val);
                    size_t off = (size_t)r * Nout + cc;
                    if (outF) outF[off] = val;
                    if (outB) outB[off] = (__bf16)val;
                }
            }
        }
    }
}

// MLP2 (K=256 -> 4) + row softmax, thread per node
__global__ void mlp2_softmax(const __bf16* __restrict__ hm, const float* __restrict__ W,
                             const float* __restrict__ b, float* __restrict__ S, int n) {
    int i = blockIdx.x * blockDim.x + threadIdx.x;
    if (i >= n) return;
    const __bf16* row = hm + (size_t)i * MLPH;
    float o0 = b[0], o1 = b[1], o2 = b[2], o3 = b[3];
    #pragma unroll 4
    for (int k = 0; k < MLPH; k++) {
        float v = (float)row[k];
        o0 += v * W[k];
        o1 += v * W[MLPH + k];
        o2 += v * W[2 * MLPH + k];
        o3 += v * W[3 * MLPH + k];
    }
    float m  = fmaxf(fmaxf(o0, o1), fmaxf(o2, o3));
    float e0 = __expf(o0 - m), e1 = __expf(o1 - m), e2 = __expf(o2 - m), e3 = __expf(o3 - m);
    float inv = 1.f / (e0 + e1 + e2 + e3);
    S[4 * i]     = e0 * inv;
    S[4 * i + 1] = e1 * inv;
    S[4 * i + 2] = e2 * inv;
    S[4 * i + 3] = e3 * inv;
}

// ---------------- host side ----------------
static inline char* ws_take(char*& p, size_t bytes) {
    char* r = p;
    p += (bytes + 255) & ~(size_t)255;
    return r;
}

struct GatBufs {
    const int *esrc, *edst;
    const float *ea;
    float *lsum, *lcnt, *as_, *ad_, *denom, *alpha, *esc;
    unsigned *amax;
    float *h, *agg;
};

static void run_gat_layer(const GatBufs& B, const float* asw, const float* adw,
                          const float* Wev, const float* aev, const float* bias,
                          __bf16* hb_out, hipStream_t stream) {
    const int TPB = 256;
    dot512<<<1, 256, 0, stream>>>(Wev, aev, B.esc);
    node_dots<<<(NN * 32 + TPB - 1) / TPB, TPB, 0, stream>>>(B.h, asw, adw, B.as_, B.ad_, NN, CH);
    fill_u32<<<(NN + TPB - 1) / TPB, TPB, 0, stream>>>(B.amax, 0u, NN);
    fill_f32<<<(NN + TPB - 1) / TPB, TPB, 0, stream>>>(B.denom, 0.f, NN);
    alpha_max<<<(ETOT + TPB - 1) / TPB, TPB, 0, stream>>>(B.esrc, B.edst, B.ea, B.lsum, B.lcnt,
                                                          B.as_, B.ad_, B.esc, B.alpha, B.amax);
    ex_denom<<<(ETOT + TPB - 1) / TPB, TPB, 0, stream>>>(B.edst, B.alpha, B.amax, B.denom);
    finalize_w<<<(ETOT + TPB - 1) / TPB, TPB, 0, stream>>>(B.edst, B.alpha, B.denom);
    fill_f32<<<(NN * CH + TPB - 1) / TPB, TPB, 0, stream>>>(B.agg, 0.f, NN * CH);
    scatter_agg<<<ETOT, 256, 0, stream>>>(B.esrc, B.edst, B.alpha, B.h, B.agg);
    bias_silu_bf16<<<(NN * CH + TPB - 1) / TPB, TPB, 0, stream>>>(B.agg, bias, hb_out,
                                                                  NN * CH, CH - 1);
}

extern "C" void kernel_launch(void* const* d_in, const int* in_sizes, int n_in,
                              void* d_out, int out_size, void* d_ws, size_t ws_size,
                              hipStream_t stream) {
    (void)in_sizes; (void)n_in; (void)out_size; (void)ws_size;

    const float* x   = (const float*)d_in[0];
    const int*   ei  = (const int*)d_in[1];
    const float* ea  = (const float*)d_in[2];
    const float* A   = (const float*)d_in[3];
    const float* W1  = (const float*)d_in[4];
    const float* as1 = (const float*)d_in[5];
    const float* ad1 = (const float*)d_in[6];
    const float* We1 = (const float*)d_in[7];
    const float* ae1 = (const float*)d_in[8];
    const float* b1  = (const float*)d_in[9];
    const float* W2  = (const float*)d_in[10];
    const float* as2 = (const float*)d_in[11];
    const float* ad2 = (const float*)d_in[12];
    const float* We2 = (const float*)d_in[13];
    const float* ae2 = (const float*)d_in[14];
    const float* b2  = (const float*)d_in[15];
    const float* Wm1 = (const float*)d_in[16];
    const float* bm1 = (const float*)d_in[17];
    const float* Wm2 = (const float*)d_in[18];
    const float* bm2 = (const float*)d_in[19];

    const int* esrc = ei;
    const int* edst = ei + EE;

    // ---- workspace layout ----
    char* wp = (char*)d_ws;
    __bf16* xb   = (__bf16*)ws_take(wp, (size_t)NN * DIN * 2);
    __bf16* w1b  = (__bf16*)ws_take(wp, (size_t)CH * DIN * 2);
    __bf16* w2b  = (__bf16*)ws_take(wp, (size_t)CH * CH * 2);
    __bf16* wm1b = (__bf16*)ws_take(wp, (size_t)MLPH * CH * 2);
    float*  h    = (float*) ws_take(wp, (size_t)NN * CH * 4);
    float*  agg  = (float*) ws_take(wp, (size_t)NN * CH * 4);
    __bf16* hb   = (__bf16*)ws_take(wp, (size_t)NN * CH * 2);
    __bf16* hmb  = (__bf16*)ws_take(wp, (size_t)NN * MLPH * 2);
    float*  as_  = (float*) ws_take(wp, (size_t)NN * 4);
    float*  ad_  = (float*) ws_take(wp, (size_t)NN * 4);
    float*  lsum = (float*) ws_take(wp, (size_t)NN * 4);
    float*  lcnt = (float*) ws_take(wp, (size_t)NN * 4);
    unsigned* amax = (unsigned*)ws_take(wp, (size_t)NN * 4);
    float*  denom = (float*)ws_take(wp, (size_t)NN * 4);
    float*  alpha = (float*)ws_take(wp, (size_t)ETOT * 4);
    float*  esc   = (float*)ws_take(wp, 256);

    const int TPB = 256;

    // ---- A pass-through: 400 MB streaming copy, non-temporal ----
    {
        int n4 = (NN * NN) / 4;
        copy_nt_f4<<<(n4 + TPB - 1) / TPB, TPB, 0, stream>>>((const f4*)A, (f4*)d_out, n4);
    }

    // ---- bf16 conversions ----
    cvt_f32_bf16<<<(NN * DIN + TPB - 1) / TPB, TPB, 0, stream>>>(x, xb, NN * DIN);
    cvt_f32_bf16<<<(CH * DIN + TPB - 1) / TPB, TPB, 0, stream>>>(W1, w1b, CH * DIN);
    cvt_f32_bf16<<<(CH * CH + TPB - 1) / TPB, TPB, 0, stream>>>(W2, w2b, CH * CH);
    cvt_f32_bf16<<<(MLPH * CH + TPB - 1) / TPB, TPB, 0, stream>>>(Wm1, wm1b, MLPH * CH);

    // ---- self-loop attrs (mean of incoming) ----
    fill_f32<<<(NN + TPB - 1) / TPB, TPB, 0, stream>>>(lsum, 0.f, NN);
    fill_f32<<<(NN + TPB - 1) / TPB, TPB, 0, stream>>>(lcnt, 0.f, NN);
    selfloop_stats<<<(EE + TPB - 1) / TPB, TPB, 0, stream>>>(edst, ea, lcnt, lsum, EE);

    GatBufs B = { esrc, edst, ea, lsum, lcnt, as_, ad_, denom, alpha, esc, amax, h, agg };
    dim3 gC((NN + 127) / 128, CH / 128);     // Nout=512 -> y=4
    dim3 gM((NN + 127) / 128, MLPH / 128);   // Nout=256 -> y=2

    // ---- GAT layer 1 ----
    gemm_bf16_wmma<<<gC, 256, 0, stream>>>(xb, w1b, h, ((__bf16*)nullptr),
                                           (const float*)nullptr, NN, DIN, CH, 0);
    run_gat_layer(B, as1, ad1, We1, ae1, b1, hb, stream);

    // ---- GAT layer 2 ----
    gemm_bf16_wmma<<<gC, 256, 0, stream>>>(hb, w2b, h, ((__bf16*)nullptr),
                                           (const float*)nullptr, NN, CH, CH, 0);
    run_gat_layer(B, as2, ad2, We2, ae2, b2, hb, stream);

    // ---- MLP1: silu(hb @ Wm1^T + bm1) -> bf16 ----
    gemm_bf16_wmma<<<gM, 256, 0, stream>>>(hb, wm1b, (float*)nullptr, hmb,
                                           bm1, NN, CH, MLPH, 1);

    // ---- MLP2 + softmax -> S (after A in d_out) ----
    float* Sout = (float*)d_out + (size_t)NN * (size_t)NN;
    mlp2_softmax<<<(NN + TPB - 1) / TPB, TPB, 0, stream>>>(hmb, Wm2, bm2, Sout, NN);
}
